// CaiT_47811575939183
// MI455X (gfx1250) — matrix operand order repaired and checked by reading, not compiled, not run on hardware
//
#include <hip/hip_runtime.h>
#include <math.h>

// ---------------------------------------------------------------------------
// CaiT forward for MI455X (gfx1250, wave32). All heavy GEMMs run through
// v_wmma_f32_16x16x32_bf16 (bf16 in, f32 accumulate). Small per-row ops
// (LayerNorm, talking-heads softmax, 1-token class attention) stay f32 VALU.
// GEMM uses double-buffered LDS: global b128 loads for tile k+1 are issued
// before the WMMAs of tile k, hiding global latency behind the matrix pipe.
// Edge tiles use clamped, branch-free (cndmask) scalar loads.
// ---------------------------------------------------------------------------

typedef __attribute__((ext_vector_type(16))) __bf16           v16bf;
typedef __attribute__((ext_vector_type(8)))  float            v8f;
typedef __attribute__((ext_vector_type(8)))  unsigned short   ushort8;

union FragBF { ushort8 u[2]; v16bf v; };

// ----- model dims -----
#define BSZ      64
#define NPATCH   196
#define DIMM     192
#define HEADS    4
#define DHEAD    64
#define INNER    256
#define MLPD     384
#define NCLS     1000
#define PATCHDIM 3840
#define NTOK     (BSZ * NPATCH)          // 12544
#define NCTX     197
#define NCTXTOK  (BSZ * NCTX)            // 12608
#define JPAD     224                     // 196 padded to mult of 32
#define MASK_VAL (-1e-9f)

// ----- helpers -----
__device__ __forceinline__ unsigned short f2bf(float f) {
  unsigned u = __float_as_uint(f);
  unsigned r = u + 0x7FFFu + ((u >> 16) & 1u);   // round to nearest even
  return (unsigned short)(r >> 16);
}
__device__ __forceinline__ float gelu_exact(float x) {
  return 0.5f * x * (1.0f + erff(x * 0.70710678118654752f));
}
__device__ __forceinline__ float blk_sum(float v, float* red) {
  int t = threadIdx.x;
  red[t] = v; __syncthreads();
  #pragma unroll
  for (int s = 128; s > 0; s >>= 1) { if (t < s) red[t] += red[t + s]; __syncthreads(); }
  float r = red[0]; __syncthreads();
  return r;
}
__device__ __forceinline__ float blk_max(float v, float* red) {
  int t = threadIdx.x;
  red[t] = v; __syncthreads();
  #pragma unroll
  for (int s = 128; s > 0; s >>= 1) { if (t < s) red[t] = fmaxf(red[t], red[t + s]); __syncthreads(); }
  float r = red[0]; __syncthreads();
  return r;
}
__device__ __forceinline__ void store_out(float* p, float v)          { *p = v; }
__device__ __forceinline__ void store_out(unsigned short* p, float v) { *p = f2bf(v); }

// ---------------------------------------------------------------------------
// Generic batched WMMA GEMM:  C[z] = epi( A[z](MxK,bf16) * B[z](KxN,bf16) )
// Tile 128(M) x 64(N) x 32(K), 256 threads = 8 waves, each wave a 32x32
// sub-tile (2x2 WMMA 16x16x32 bf16). Double-buffered LDS ping-pong:
//   stage(kt+1) -> regs  |  WMMA on LDS[kt&1]  |  sync; regs -> LDS[~kt&1]
// Batch z = (zb, zh) = (z/nh, z%nh) with independent strides (batch, head).
// TRANSB: B supplied row-major [N,K] (used for Q*K^T).
// EPI: 0 none, 1 +bias, 2 +bias then exact GELU.  bscale: per-(z%smod) scalar.
// Fast-path alignment contract: all lda/ldb are multiples of 8 elements,
// k0 multiple of 32, so every uint4 access is 16B-aligned.
// ---------------------------------------------------------------------------
#define TM 128
#define TN 64
#define TK 32
#define LSTR 48   // LDS row stride (bf16 elems); 96B rows keep b128 alignment

// register-staged tile loads (shared by fast + branch-free edge paths)
__device__ __forceinline__ void load_a_stage(
    const unsigned short* __restrict__ Ab, int lda, int M, int K,
    int gm, int k0, int a_cb, bool fast, uint4& v0, uint4& v1)
{
  if (fast) {
    const uint4* src = (const uint4*)(Ab + (long long)gm * lda + k0 + a_cb);
    v0 = src[0];
    v1 = src[1];
  } else {
    unsigned short tmp[16];
    const int gmc = gm < M ? gm : (M - 1);
    const bool mok = gm < M;
    #pragma unroll
    for (int i = 0; i < 16; i++) {
      const int gk = k0 + a_cb + i;
      const int gkc = gk < K ? gk : (K - 1);
      unsigned short v = Ab[(long long)gmc * lda + gkc];   // clamped: in-bounds
      tmp[i] = (mok && gk < K) ? v : (unsigned short)0;
    }
    v0 = *(const uint4*)&tmp[0];
    v1 = *(const uint4*)&tmp[8];
  }
}

template <bool TRANSB>
__device__ __forceinline__ void load_b_stage(
    const unsigned short* __restrict__ Bb, int ldb, int N, int K,
    int k0, int n0, int bt_n, int bt_kb, int bn_k, int bn_nb,
    bool fast, uint4& v)
{
  if (TRANSB) {
    const int gn  = n0 + bt_n;
    const int gk0 = k0 + bt_kb;
    if (fast) {
      v = *(const uint4*)(Bb + (long long)gn * ldb + gk0);
    } else {
      unsigned short tmp[8];
      const int gnc = gn < N ? gn : (N - 1);
      const bool nok = gn < N;
      #pragma unroll
      for (int i = 0; i < 8; i++) {
        const int gk = gk0 + i;
        const int gkc = gk < K ? gk : (K - 1);
        unsigned short w = Bb[(long long)gnc * ldb + gkc];
        tmp[i] = (nok && gk < K) ? w : (unsigned short)0;
      }
      v = *(const uint4*)&tmp[0];
    }
  } else {
    const int gk = k0 + bn_k;
    const int gn = n0 + bn_nb;
    if (fast) {
      v = *(const uint4*)(Bb + (long long)gk * ldb + gn);
    } else {
      unsigned short tmp[8];
      const int gkc = gk < K ? gk : (K - 1);
      const bool kok = gk < K;
      #pragma unroll
      for (int i = 0; i < 8; i++) {
        const int gni = gn + i;
        const int gnc = gni < N ? gni : (N - 1);
        unsigned short w = Bb[(long long)gkc * ldb + gnc];
        tmp[i] = (kok && gni < N) ? w : (unsigned short)0;
      }
      v = *(const uint4*)&tmp[0];
    }
  }
}

template <typename OutT, int EPI, bool TRANSB>
__global__ __launch_bounds__(256) void gemm_wmma_kernel(
    const unsigned short* __restrict__ A, long long sAb, long long sAh, int lda,
    const unsigned short* __restrict__ B, long long sBb, long long sBh, int ldb,
    OutT* __restrict__ C, long long sCb, long long sCh, int ldc,
    const float* __restrict__ bias, const float* __restrict__ bscale, int smod,
    int M, int N, int K, int nh)
{
  __shared__ __align__(16) unsigned short As[2][TM * LSTR];
  __shared__ __align__(16) unsigned short Bs[2][TN * LSTR];

  const int z  = blockIdx.z;
  const int zb = z / nh, zh = z % nh;
  const unsigned short* Ab = A + (long long)zb * sAb + (long long)zh * sAh;
  const unsigned short* Bb = B + (long long)zb * sBb + (long long)zh * sBh;
  OutT* Cb = C + (long long)zb * sCb + (long long)zh * sCh;

  const int m0 = blockIdx.y * TM;
  const int n0 = blockIdx.x * TN;
  const bool mfull = (m0 + TM) <= M;
  const bool nfull = (n0 + TN) <= N;
  const int tid  = threadIdx.x;
  const int wave = tid >> 5;
  const int lane = tid & 31;
  const int wm = wave >> 1;       // 0..3 : 32-row band
  const int wn = wave & 1;        // 0..1 : 32-col band
  const int half = lane >> 4;
  const int r    = lane & 15;

  // staging coordinates
  const int a_row = tid >> 1;            // 0..127
  const int a_cb  = (tid & 1) * 16;      // 0 or 16
  const int gm_ld = m0 + a_row;
  const int bt_n  = tid & 63;            // TRANSB: 0..63
  const int bt_kb = (tid >> 6) * 8;      // TRANSB: 0,8,16,24
  const int bn_k  = tid >> 3;            // !TRANSB: 0..31
  const int bn_nb = (tid & 7) * 8;       // !TRANSB: 0..56

  v8f acc[2][2];
  #pragma unroll
  for (int i = 0; i < 2; i++)
    #pragma unroll
    for (int j = 0; j < 2; j++)
      #pragma unroll
      for (int e = 0; e < 8; e++) acc[i][j][e] = 0.0f;

  const int nk = (K + TK - 1) / TK;

  // prologue: stage tile 0 into LDS buffer 0
  {
    const bool kf = TK <= K;
    uint4 a0, a1, b0;
    load_a_stage(Ab, lda, M, K, gm_ld, 0, a_cb, mfull && kf, a0, a1);
    load_b_stage<TRANSB>(Bb, ldb, N, K, 0, n0, bt_n, bt_kb, bn_k, bn_nb, nfull && kf, b0);
    *(uint4*)&As[0][a_row * LSTR + a_cb]     = a0;
    *(uint4*)&As[0][a_row * LSTR + a_cb + 8] = a1;
    if (TRANSB) {
      *(uint4*)&Bs[0][bt_n * LSTR + bt_kb] = b0;
    } else {
      const unsigned short* pv = (const unsigned short*)&b0;
      #pragma unroll
      for (int i = 0; i < 8; i++) Bs[0][(bn_nb + i) * LSTR + bn_k] = pv[i];
    }
  }
  __syncthreads();

  for (int kt = 0; kt < nk; kt++) {
    const int cur = kt & 1;
    const bool more = (kt + 1) < nk;

    // issue global loads for tile kt+1 BEFORE the WMMAs of tile kt
    uint4 na0, na1, nb0;
    if (more) {
      const int k0n = (kt + 1) * TK;
      const bool kf = (k0n + TK) <= K;
      load_a_stage(Ab, lda, M, K, gm_ld, k0n, a_cb, mfull && kf, na0, na1);
      load_b_stage<TRANSB>(Bb, ldb, N, K, k0n, n0, bt_n, bt_kb, bn_k, bn_nb, nfull && kf, nb0);
    }

    // ---- compute tile kt from LDS[cur] ----
    FragBF af[2], bf[2];
    #pragma unroll
    for (int tm = 0; tm < 2; tm++) {
      const int row = wm * 32 + tm * 16 + r;
      af[tm].u[0] = *(const ushort8*)&As[cur][row * LSTR + half * 8];
      af[tm].u[1] = *(const ushort8*)&As[cur][row * LSTR + 16 + half * 8];
    }
    #pragma unroll
    for (int tn = 0; tn < 2; tn++) {
      const int col = wn * 32 + tn * 16 + r;
      bf[tn].u[0] = *(const ushort8*)&Bs[cur][col * LSTR + half * 8];
      bf[tn].u[1] = *(const ushort8*)&Bs[cur][col * LSTR + 16 + half * 8];
    }
    #pragma unroll
    for (int tm = 0; tm < 2; tm++)
      #pragma unroll
      for (int tn = 0; tn < 2; tn++)
        acc[tm][tn] = __builtin_amdgcn_wmma_f32_16x16x32_bf16(
            false, af[tm].v, false, bf[tn].v, (short)0, acc[tm][tn], false, false);

    // ---- commit staged registers into the other LDS buffer ----
    if (more) {
      const int nxt = cur ^ 1;
      __syncthreads();   // everyone finished reading LDS[nxt] (tile kt-1)
      *(uint4*)&As[nxt][a_row * LSTR + a_cb]     = na0;
      *(uint4*)&As[nxt][a_row * LSTR + a_cb + 8] = na1;
      if (TRANSB) {
        *(uint4*)&Bs[nxt][bt_n * LSTR + bt_kb] = nb0;
      } else {
        const unsigned short* pv = (const unsigned short*)&nb0;
        #pragma unroll
        for (int i = 0; i < 8; i++) Bs[nxt][(bn_nb + i) * LSTR + bn_k] = pv[i];
      }
      __syncthreads();   // LDS[nxt] ready for tile kt+1
    }
  }

  // ---- epilogue ----
  const float scl = (bscale != nullptr) ? bscale[z % smod] : 1.0f;
  #pragma unroll
  for (int tm = 0; tm < 2; tm++) {
    #pragma unroll
    for (int tn = 0; tn < 2; tn++) {
      const int rbase = m0 + wm * 32 + tm * 16 + half * 8;
      const int col   = n0 + wn * 32 + tn * 16 + r;
      if (col < N) {
        #pragma unroll
        for (int g = 0; g < 8; g++) {
          const int row = rbase + g;
          if (row < M) {
            float v = acc[tm][tn][g] * scl;
            if (EPI >= 1) v += bias[col];
            if (EPI == 2) v = gelu_exact(v);
            store_out(&Cb[(long long)row * ldc + col], v);
          }
        }
      }
    }
  }
}

// ---------------------------------------------------------------------------
// SPM: patch shifting gather + LayerNorm(3840) -> bf16 rows for the merge GEMM
// ---------------------------------------------------------------------------
__global__ __launch_bounds__(256) void spm_kernel(
    const float* __restrict__ img, const float* __restrict__ g,
    const float* __restrict__ bb, unsigned short* __restrict__ Xp)
{
  __shared__ float vals[PATCHDIM];
  __shared__ float red[256];
  const int id  = blockIdx.x;          // 0..12543
  const int bI  = id / NPATCH;
  const int pid = id % NPATCH;
  const int ph = pid / 14, pw = pid % 14;
  const int t = threadIdx.x;
  const int dy[5] = {0, -8, -8, 8, 8};
  const int dx[5] = {0, -8, 8, -8, 8};

  float s = 0.0f, s2 = 0.0f;
  for (int f = t; f < PATCHDIM; f += 256) {
    const int pp = f / 15, ch = f % 15;
    const int p1 = pp / 16, p2 = pp % 16;
    const int grp = ch / 3, c = ch % 3;
    const int y = ph * 16 + p1 + dy[grp];
    const int x = pw * 16 + p2 + dx[grp];
    float v = 0.0f;
    if (y >= 0 && y < 224 && x >= 0 && x < 224)
      v = img[(((long long)bI * 3 + c) * 224 + y) * 224 + x];
    vals[f] = v; s += v; s2 += v * v;
  }
  const float mean = blk_sum(s, red)  * (1.0f / PATCHDIM);
  const float ex2  = blk_sum(s2, red) * (1.0f / PATCHDIM);
  const float rstd = rsqrtf(ex2 - mean * mean + 1e-5f);
  __syncthreads();
  for (int f = t; f < PATCHDIM; f += 256)
    Xp[(long long)id * PATCHDIM + f] = f2bf((vals[f] - mean) * rstd * g[f] + bb[f]);
}

// ---------------------------------------------------------------------------
// LayerNorm over rows of `cols` (<=256), bf16 out with arbitrary row stride
// ---------------------------------------------------------------------------
__global__ __launch_bounds__(256) void ln_rows_kernel(
    const float* __restrict__ x, const float* __restrict__ g,
    const float* __restrict__ bb, unsigned short* __restrict__ out,
    long long ostride, int cols)
{
  __shared__ float red[256];
  const int row = blockIdx.x;
  const int c = threadIdx.x;
  const bool act = c < cols;
  const float v = act ? x[(long long)row * cols + c] : 0.0f;
  const float mean = blk_sum(v, red) / cols;
  const float dv = act ? (v - mean) : 0.0f;
  const float var = blk_sum(dv * dv, red) / cols;
  const float rstd = rsqrtf(var + 1e-5f);
  if (act) out[row * ostride + c] = f2bf(dv * rstd * g[c] + bb[c]);
}

// ---------------------------------------------------------------------------
// Patch attention: diag mask -> talking-heads pre -> softmax -> post -> bf16
// One block per (b, i) query row; thread j handles one key column.
// ---------------------------------------------------------------------------
__global__ __launch_bounds__(256) void talking_softmax_kernel(
    const float* __restrict__ dots, unsigned short* __restrict__ attnb,
    const float* __restrict__ mpre, const float* __restrict__ mpost)
{
  __shared__ float red[256];
  const int blk = blockIdx.x;
  const int b = blk / NPATCH;
  const int i = blk % NPATCH;
  const int j = threadIdx.x;
  const bool act = j < NPATCH;

  float d[HEADS];
  const long long base = ((long long)b * HEADS) * NPATCH * JPAD + (long long)i * JPAD + j;
  #pragma unroll
  for (int h = 0; h < HEADS; h++)
    d[h] = act ? dots[base + (long long)h * NPATCH * JPAD] : 0.0f;
  if (act && j == i) {
    #pragma unroll
    for (int h = 0; h < HEADS; h++) d[h] = MASK_VAL;
  }
  float m[HEADS];
  #pragma unroll
  for (int gq = 0; gq < HEADS; gq++) {
    float acc = 0.0f;
    #pragma unroll
    for (int h = 0; h < HEADS; h++) acc += d[h] * mpre[h * HEADS + gq];
    m[gq] = acc;
  }
  float sm[HEADS];
  #pragma unroll
  for (int gq = 0; gq < HEADS; gq++) {
    const float mx = blk_max(act ? m[gq] : -3.4e38f, red);
    const float e  = act ? __expf(m[gq] - mx) : 0.0f;
    const float s  = blk_sum(e, red);
    sm[gq] = e / s;
  }
  if (act) {
    #pragma unroll
    for (int gq = 0; gq < HEADS; gq++) {
      float acc = 0.0f;
      #pragma unroll
      for (int h = 0; h < HEADS; h++) acc += sm[h] * mpost[h * HEADS + gq];
      attnb[((long long)(b * HEADS + gq) * NPATCH + i) * JPAD + j] = f2bf(acc);
    }
  }
}

// ---------------------------------------------------------------------------
// Class attention: 1 query token, 197 keys; all-VALU (tiny). One block per b.
// ---------------------------------------------------------------------------
__global__ __launch_bounds__(256) void cls_attn_kernel(
    const float* __restrict__ qc, const float* __restrict__ kvc,
    const float* __restrict__ scale, const float* __restrict__ mpre,
    const float* __restrict__ mpost, unsigned short* __restrict__ acls)
{
  __shared__ float qs[INNER];
  __shared__ float smv[HEADS * NCTX];
  __shared__ float red[256];
  const int b = blockIdx.x;
  const int t = threadIdx.x;
  qs[t] = qc[b * INNER + t];
  __syncthreads();

  const int j = t;
  const bool act = j < NCTX;
  float d[HEADS];
  if (act) {
    const float* kr = kvc + ((long long)(b * NCTX + j)) * (2 * INNER);
    #pragma unroll
    for (int h = 0; h < HEADS; h++) {
      float s = 0.0f;
      for (int dd = 0; dd < DHEAD; dd++) s += qs[h * DHEAD + dd] * kr[h * DHEAD + dd];
      d[h] = s * scale[h];
    }
    if (j == 0) {
      #pragma unroll
      for (int h = 0; h < HEADS; h++) d[h] = MASK_VAL;
    }
  } else {
    #pragma unroll
    for (int h = 0; h < HEADS; h++) d[h] = 0.0f;
  }
  float m[HEADS];
  #pragma unroll
  for (int gq = 0; gq < HEADS; gq++) {
    float acc = 0.0f;
    #pragma unroll
    for (int h = 0; h < HEADS; h++) acc += d[h] * mpre[h * HEADS + gq];
    m[gq] = acc;
  }
  float sm[HEADS];
  #pragma unroll
  for (int gq = 0; gq < HEADS; gq++) {
    const float mx = blk_max(act ? m[gq] : -3.4e38f, red);
    const float e  = act ? __expf(m[gq] - mx) : 0.0f;
    const float s  = blk_sum(e, red);
    sm[gq] = e / s;
  }
  float a[HEADS];
  #pragma unroll
  for (int gq = 0; gq < HEADS; gq++) {
    float acc = 0.0f;
    #pragma unroll
    for (int h = 0; h < HEADS; h++) acc += sm[h] * mpost[h * HEADS + gq];
    a[gq] = acc;
  }
  if (act) {
    #pragma unroll
    for (int gq = 0; gq < HEADS; gq++) smv[gq * NCTX + j] = a[gq];
  }
  __syncthreads();

  // out[g,d] = sum_j attn[g,j] * v[j, g*64+d]
  const int gq = t >> 6;
  const int dd = t & 63;
  float o = 0.0f;
  for (int jj = 0; jj < NCTX; jj++)
    o += smv[gq * NCTX + jj] *
         kvc[((long long)(b * NCTX + jj)) * (2 * INNER) + INNER + gq * DHEAD + dd];
  acls[b * INNER + t] = f2bf(o);
}

// ---------------------------------------------------------------------------
// Elementwise utilities
// ---------------------------------------------------------------------------
__global__ __launch_bounds__(256) void f32_to_bf16_kernel(
    const float* __restrict__ in, unsigned short* __restrict__ out, long long n)
{
  long long i = (long long)blockIdx.x * 256 + threadIdx.x;
  if (i < n) out[i] = f2bf(in[i]);
}
__global__ __launch_bounds__(256) void residual_ls_kernel(
    float* __restrict__ x, const float* __restrict__ y,
    const float* __restrict__ ls, long long n, int cols)
{
  long long i = (long long)blockIdx.x * 256 + threadIdx.x;
  if (i < n) x[i] += y[i] * ls[i % cols];
}
__global__ __launch_bounds__(256) void add_mod_kernel(
    float* __restrict__ x, const float* __restrict__ a, long long n, int mod)
{
  long long i = (long long)blockIdx.x * 256 + threadIdx.x;
  if (i < n) x[i] += a[i % mod];
}
__global__ __launch_bounds__(256) void bcast_kernel(
    float* __restrict__ x, const float* __restrict__ a, long long n, int mod)
{
  long long i = (long long)blockIdx.x * 256 + threadIdx.x;
  if (i < n) x[i] = a[i % mod];
}
__global__ __launch_bounds__(256) void to_ctx_kernel(
    const float* __restrict__ x, unsigned short* __restrict__ ctx, long long n)
{
  long long i = (long long)blockIdx.x * 256 + threadIdx.x;
  if (i < n) {
    const int m = (int)(i / DIMM), c = (int)(i % DIMM);
    const int b = m / NPATCH, j = m % NPATCH;
    ctx[((long long)(b * NCTX + 1 + j)) * DIMM + c] = f2bf(x[i]);
  }
}

// ---------------------------------------------------------------------------
// Host-side launch helpers
// ---------------------------------------------------------------------------
static inline int cdiv(int a, int b) { return (a + b - 1) / b; }

template <typename OutT, int EPI, bool TRANSB>
static void gemm(hipStream_t st,
                 const unsigned short* A, long long sAb, long long sAh, int lda,
                 const unsigned short* B, long long sBb, long long sBh, int ldb,
                 OutT* C, long long sCb, long long sCh, int ldc,
                 const float* bias, const float* bscale, int smod,
                 int M, int N, int K, int batch, int nh)
{
  dim3 gr(cdiv(N, TN), cdiv(M, TM), batch);
  gemm_wmma_kernel<OutT, EPI, TRANSB><<<gr, 256, 0, st>>>(
      A, sAb, sAh, lda, B, sBb, sBh, ldb, C, sCb, sCh, ldc,
      bias, bscale, smod, M, N, K, nh);
}

static void conv_bf16(hipStream_t st, const float* in, unsigned short* out, long long n) {
  f32_to_bf16_kernel<<<(unsigned)((n + 255) / 256), 256, 0, st>>>(in, out, n);
}

extern "C" void kernel_launch(void* const* d_in, const int* in_sizes, int n_in,
                              void* d_out, int out_size, void* d_ws, size_t ws_size,
                              hipStream_t stream) {
  (void)in_sizes; (void)n_in; (void)out_size; (void)ws_size;
  typedef unsigned short bfx;

  // ---- inputs (setup_inputs dict order) ----
  const float* img      = (const float*)d_in[0];
  const float* ln_m_g   = (const float*)d_in[1];
  const float* ln_m_b   = (const float*)d_in[2];
  const float* w_merge  = (const float*)d_in[3];
  const float* pos_emb  = (const float*)d_in[4];
  const float* cls_tok  = (const float*)d_in[5];
  // pt: 6..22
  const float* pt_ln1g = (const float*)d_in[6];
  const float* pt_ln1b = (const float*)d_in[7];
  const float* pt_wq   = (const float*)d_in[8];
  const float* pt_wkv  = (const float*)d_in[9];
  const float* pt_sc   = (const float*)d_in[10];
  const float* pt_mpre = (const float*)d_in[11];
  const float* pt_mpost= (const float*)d_in[12];
  const float* pt_wo   = (const float*)d_in[13];
  const float* pt_bo   = (const float*)d_in[14];
  const float* pt_ls1  = (const float*)d_in[15];
  const float* pt_ln2g = (const float*)d_in[16];
  const float* pt_ln2b = (const float*)d_in[17];
  const float* pt_w1   = (const float*)d_in[18];
  const float* pt_b1   = (const float*)d_in[19];
  const float* pt_w2   = (const float*)d_in[20];
  const float* pt_b2   = (const float*)d_in[21];
  const float* pt_ls2  = (const float*)d_in[22];
  // ct: 23..39
  const float* ct_ln1g = (const float*)d_in[23];
  const float* ct_ln1b = (const float*)d_in[24];
  const float* ct_wq   = (const float*)d_in[25];
  const float* ct_wkv  = (const float*)d_in[26];
  const float* ct_sc   = (const float*)d_in[27];
  const float* ct_mpre = (const float*)d_in[28];
  const float* ct_mpost= (const float*)d_in[29];
  const float* ct_wo   = (const float*)d_in[30];
  const float* ct_bo   = (const float*)d_in[31];
  const float* ct_ls1  = (const float*)d_in[32];
  const float* ct_ln2g = (const float*)d_in[33];
  const float* ct_ln2b = (const float*)d_in[34];
  const float* ct_w1   = (const float*)d_in[35];
  const float* ct_b1   = (const float*)d_in[36];
  const float* ct_w2   = (const float*)d_in[37];
  const float* ct_b2   = (const float*)d_in[38];
  const float* ct_ls2  = (const float*)d_in[39];
  const float* head_ln_g = (const float*)d_in[40];
  const float* head_ln_b = (const float*)d_in[41];
  const float* head_w    = (const float*)d_in[42];
  const float* head_b    = (const float*)d_in[43];
  float* out = (float*)d_out;

  // ---- workspace arena ----
  char* base = (char*)d_ws;
  size_t off = 0;
  auto alloc = [&](size_t bytes) -> char* {
    char* p = base + off;
    off = (off + bytes + 255) & ~(size_t)255;
    return p;
  };
  bfx*  wmerge_h = (bfx*)alloc((size_t)PATCHDIM * DIMM * 2);
  bfx*  ptwq_h   = (bfx*)alloc((size_t)24 * DIMM * INNER * 2);
  bfx*  ptwkv_h  = (bfx*)alloc((size_t)24 * DIMM * 2 * INNER * 2);
  bfx*  ptwo_h   = (bfx*)alloc((size_t)24 * INNER * DIMM * 2);
  bfx*  ptw1_h   = (bfx*)alloc((size_t)24 * DIMM * MLPD * 2);
  bfx*  ptw2_h   = (bfx*)alloc((size_t)24 * MLPD * DIMM * 2);
  bfx*  ctwq_h   = (bfx*)alloc((size_t)2 * DIMM * INNER * 2);
  bfx*  ctwkv_h  = (bfx*)alloc((size_t)2 * DIMM * 2 * INNER * 2);
  bfx*  ctwo_h   = (bfx*)alloc((size_t)2 * INNER * DIMM * 2);
  bfx*  ctw1_h   = (bfx*)alloc((size_t)2 * DIMM * MLPD * 2);
  bfx*  ctw2_h   = (bfx*)alloc((size_t)2 * MLPD * DIMM * 2);
  bfx*  headw_h  = (bfx*)alloc((size_t)DIMM * NCLS * 2);
  bfx*  Xp    = (bfx*)alloc((size_t)NTOK * PATCHDIM * 2);
  float* x    = (float*)alloc((size_t)NTOK * DIMM * 4);
  bfx*  xn    = (bfx*)alloc((size_t)NTOK * DIMM * 2);
  bfx*  qb    = (bfx*)alloc((size_t)NTOK * INNER * 2);
  bfx*  kvb   = (bfx*)alloc((size_t)NTOK * 2 * INNER * 2);
  float* dots = (float*)alloc((size_t)BSZ * HEADS * NPATCH * JPAD * 4);
  bfx*  attnb = (bfx*)alloc((size_t)BSZ * HEADS * NPATCH * JPAD * 2);
  bfx*  ao    = (bfx*)alloc((size_t)NTOK * INNER * 2);
  float* tmpf = (float*)alloc((size_t)NTOK * DIMM * 4);
  bfx*  hbuf  = (bfx*)alloc((size_t)NTOK * MLPD * 2);
  bfx*  ctx   = (bfx*)alloc((size_t)NCTXTOK * DIMM * 2);
  float* qc   = (float*)alloc((size_t)BSZ * INNER * 4);
  float* kvc  = (float*)alloc((size_t)NCTXTOK * 2 * INNER * 4);
  bfx*  acls  = (bfx*)alloc((size_t)BSZ * INNER * 2);
  float* cls  = (float*)alloc((size_t)BSZ * DIMM * 4);
  bfx*  clsn  = (bfx*)alloc((size_t)BSZ * DIMM * 2);
  float* tmpc = (float*)alloc((size_t)BSZ * DIMM * 4);
  bfx*  hc    = (bfx*)alloc((size_t)BSZ * MLPD * 2);

  // ---- weights f32 -> bf16 (cheap; keeps launch deterministic) ----
  conv_bf16(stream, w_merge, wmerge_h, (long long)PATCHDIM * DIMM);
  conv_bf16(stream, pt_wq,   ptwq_h,   (long long)24 * DIMM * INNER);
  conv_bf16(stream, pt_wkv,  ptwkv_h,  (long long)24 * DIMM * 2 * INNER);
  conv_bf16(stream, pt_wo,   ptwo_h,   (long long)24 * INNER * DIMM);
  conv_bf16(stream, pt_w1,   ptw1_h,   (long long)24 * DIMM * MLPD);
  conv_bf16(stream, pt_w2,   ptw2_h,   (long long)24 * MLPD * DIMM);
  conv_bf16(stream, ct_wq,   ctwq_h,   (long long)2 * DIMM * INNER);
  conv_bf16(stream, ct_wkv,  ctwkv_h,  (long long)2 * DIMM * 2 * INNER);
  conv_bf16(stream, ct_wo,   ctwo_h,   (long long)2 * INNER * DIMM);
  conv_bf16(stream, ct_w1,   ctw1_h,   (long long)2 * DIMM * MLPD);
  conv_bf16(stream, ct_w2,   ctw2_h,   (long long)2 * MLPD * DIMM);
  conv_bf16(stream, head_w,  headw_h,  (long long)DIMM * NCLS);

  // ---- SPM: shift-gather + LN(3840) -> bf16, then merge GEMM, + pos ----
  spm_kernel<<<NTOK, 256, 0, stream>>>(img, ln_m_g, ln_m_b, Xp);
  gemm<float, 0, false>(stream, Xp, 0, 0, PATCHDIM, wmerge_h, 0, 0, DIMM,
                        x, 0, 0, DIMM, nullptr, nullptr, 1,
                        NTOK, DIMM, PATCHDIM, 1, 1);
  add_mod_kernel<<<cdiv(NTOK * DIMM, 256), 256, 0, stream>>>(
      x, pos_emb, (long long)NTOK * DIMM, NPATCH * DIMM);

  const long long sQb = (long long)NPATCH * INNER;       // q batch stride (b)
  const long long sKVb = (long long)NPATCH * 2 * INNER;  // kv batch stride (b)
  const long long sDot = (long long)NPATCH * JPAD;       // dots per-(b,h)

  // ---- patch transformer: 24 LMSA layers ----
  for (int l = 0; l < 24; l++) {
    const bfx* wq_l  = ptwq_h  + (long long)l * DIMM * INNER;
    const bfx* wkv_l = ptwkv_h + (long long)l * DIMM * 2 * INNER;
    const bfx* wo_l  = ptwo_h  + (long long)l * INNER * DIMM;
    const bfx* w1_l  = ptw1_h  + (long long)l * DIMM * MLPD;
    const bfx* w2_l  = ptw2_h  + (long long)l * MLPD * DIMM;

    ln_rows_kernel<<<NTOK, 256, 0, stream>>>(x, pt_ln1g + l * DIMM, pt_ln1b + l * DIMM,
                                             xn, DIMM, DIMM);
    gemm<bfx, 0, false>(stream, xn, 0, 0, DIMM, wq_l, 0, 0, INNER,
                        qb, 0, 0, INNER, nullptr, nullptr, 1,
                        NTOK, INNER, DIMM, 1, 1);
    gemm<bfx, 0, false>(stream, xn, 0, 0, DIMM, wkv_l, 0, 0, 2 * INNER,
                        kvb, 0, 0, 2 * INNER, nullptr, nullptr, 1,
                        NTOK, 2 * INNER, DIMM, 1, 1);
    // scores[b,h] = scale[h] * q @ k^T   (batched, B row-major [N,K] = k)
    gemm<float, 0, true>(stream, qb, sQb, DHEAD, INNER,
                         kvb, sKVb, DHEAD, 2 * INNER,
                         dots, (long long)HEADS * sDot, sDot, JPAD,
                         nullptr, pt_sc + l * HEADS, HEADS,
                         NPATCH, NPATCH, DHEAD, BSZ * HEADS, HEADS);
    talking_softmax_kernel<<<NTOK, 256, 0, stream>>>(dots, attnb,
                                                     pt_mpre + l * 16, pt_mpost + l * 16);
    // out[b,h] = attn @ v
    gemm<bfx, 0, false>(stream, attnb, (long long)HEADS * sDot, sDot, JPAD,
                        kvb + INNER, sKVb, DHEAD, 2 * INNER,
                        ao, sQb, DHEAD, INNER, nullptr, nullptr, 1,
                        NPATCH, DHEAD, NPATCH, BSZ * HEADS, HEADS);
    gemm<float, 1, false>(stream, ao, 0, 0, INNER, wo_l, 0, 0, DIMM,
                          tmpf, 0, 0, DIMM, pt_bo + l * DIMM, nullptr, 1,
                          NTOK, DIMM, INNER, 1, 1);
    residual_ls_kernel<<<cdiv(NTOK * DIMM, 256), 256, 0, stream>>>(
        x, tmpf, pt_ls1 + l * DIMM, (long long)NTOK * DIMM, DIMM);

    ln_rows_kernel<<<NTOK, 256, 0, stream>>>(x, pt_ln2g + l * DIMM, pt_ln2b + l * DIMM,
                                             xn, DIMM, DIMM);
    gemm<bfx, 2, false>(stream, xn, 0, 0, DIMM, w1_l, 0, 0, MLPD,
                        hbuf, 0, 0, MLPD, pt_b1 + l * MLPD, nullptr, 1,
                        NTOK, MLPD, DIMM, 1, 1);
    gemm<float, 1, false>(stream, hbuf, 0, 0, MLPD, w2_l, 0, 0, DIMM,
                          tmpf, 0, 0, DIMM, pt_b2 + l * DIMM, nullptr, 1,
                          NTOK, DIMM, MLPD, 1, 1);
    residual_ls_kernel<<<cdiv(NTOK * DIMM, 256), 256, 0, stream>>>(
        x, tmpf, pt_ls2 + l * DIMM, (long long)NTOK * DIMM, DIMM);
  }

  // ---- class-attention transformer ----
  to_ctx_kernel<<<cdiv(NTOK * DIMM, 256), 256, 0, stream>>>(x, ctx, (long long)NTOK * DIMM);
  bcast_kernel<<<cdiv(BSZ * DIMM, 256), 256, 0, stream>>>(cls, cls_tok,
                                                          (long long)BSZ * DIMM, DIMM);
  for (int l = 0; l < 2; l++) {
    const bfx* wq_l  = ctwq_h  + (long long)l * DIMM * INNER;
    const bfx* wkv_l = ctwkv_h + (long long)l * DIMM * 2 * INNER;
    const bfx* wo_l  = ctwo_h  + (long long)l * INNER * DIMM;
    const bfx* w1_l  = ctw1_h  + (long long)l * DIMM * MLPD;
    const bfx* w2_l  = ctw2_h  + (long long)l * MLPD * DIMM;

    // LN of cls written into ctx row 0 of each batch (row stride 197*192)
    ln_rows_kernel<<<BSZ, 256, 0, stream>>>(cls, ct_ln1g + l * DIMM, ct_ln1b + l * DIMM,
                                            ctx, (long long)NCTX * DIMM, DIMM);
    gemm<float, 0, false>(stream, ctx, 0, 0, NCTX * DIMM, wq_l, 0, 0, INNER,
                          qc, 0, 0, INNER, nullptr, nullptr, 1,
                          BSZ, INNER, DIMM, 1, 1);
    gemm<float, 0, false>(stream, ctx, 0, 0, DIMM, wkv_l, 0, 0, 2 * INNER,
                          kvc, 0, 0, 2 * INNER, nullptr, nullptr, 1,
                          NCTXTOK, 2 * INNER, DIMM, 1, 1);
    cls_attn_kernel<<<BSZ, 256, 0, stream>>>(qc, kvc, ct_sc + l * HEADS,
                                             ct_mpre + l * 16, ct_mpost + l * 16, acls);
    gemm<float, 1, false>(stream, acls, 0, 0, INNER, wo_l, 0, 0, DIMM,
                          tmpc, 0, 0, DIMM, ct_bo + l * DIMM, nullptr, 1,
                          BSZ, DIMM, INNER, 1, 1);
    residual_ls_kernel<<<cdiv(BSZ * DIMM, 256), 256, 0, stream>>>(
        cls, tmpc, ct_ls1 + l * DIMM, (long long)BSZ * DIMM, DIMM);

    ln_rows_kernel<<<BSZ, 256, 0, stream>>>(cls, ct_ln2g + l * DIMM, ct_ln2b + l * DIMM,
                                            clsn, DIMM, DIMM);
    gemm<bfx, 2, false>(stream, clsn, 0, 0, DIMM, w1_l, 0, 0, MLPD,
                        hc, 0, 0, MLPD, ct_b1 + l * MLPD, nullptr, 1,
                        BSZ, MLPD, DIMM, 1, 1);
    gemm<float, 1, false>(stream, hc, 0, 0, MLPD, w2_l, 0, 0, DIMM,
                          tmpc, 0, 0, DIMM, ct_b2 + l * DIMM, nullptr, 1,
                          BSZ, DIMM, MLPD, 1, 1);
    residual_ls_kernel<<<cdiv(BSZ * DIMM, 256), 256, 0, stream>>>(
        cls, tmpc, ct_ls2 + l * DIMM, (long long)BSZ * DIMM, DIMM);
  }

  // ---- head: LN + [192,1000] GEMM + bias -> d_out (f32 [64,1000]) ----
  ln_rows_kernel<<<BSZ, 256, 0, stream>>>(cls, head_ln_g, head_ln_b, clsn, DIMM, DIMM);
  gemm<float, 1, false>(stream, clsn, 0, 0, DIMM, headw_h, 0, 0, NCLS,
                        out, 0, 0, NCLS, head_b, nullptr, 1,
                        BSZ, NCLS, DIMM, 1, 1);
}
